// WindowAttention_48146583388335
// MI455X (gfx1250) — matrix operand order repaired and checked by reading, not compiled, not run on hardware
//
#include <hip/hip_runtime.h>
#include <hip/hip_bf16.h>

typedef __bf16 bf16_t;
typedef __attribute__((ext_vector_type(16))) __bf16 v16bf;
typedef __attribute__((ext_vector_type(8)))  float  v8f;
typedef unsigned int u32;
typedef __attribute__((ext_vector_type(4))) u32 u32x4;
typedef __attribute__((ext_vector_type(4))) int i32x4;
typedef __attribute__((ext_vector_type(8))) int i32x8;

#define TOK   49
#define HEADS 12
#define HD    32
#define DIM   384
#define DIM3  1152
#define MP    64            // padded tokens per window (4 x 16)
#define SX    392           // bf16 row stride for [64][384] LDS tiles (784B = 196 dwords, conflict-free)
#define SV    72            // bf16 row stride for v^T [384][64]
#define SP    72            // bf16 row stride for P [H*64][64]
#define QSCALE 0.17677669529663687f   // 32^-0.5

// ---- WMMA fragment helpers (contiguous-K layout used by this file for both A and B) ----
// A (16-bit, 16x32): lane L holds row M=L&15; K = (L>=16?16:0)+j, j=0..15 -> 32 contiguous bytes.
//   elements 0..7  = K = ko+0..7      ; elements 8..15 = K = ko+8..15 (at +16B)
static __device__ __forceinline__ v16bf ldfragA(const bf16_t* p) {
  union { v16bf v; u32x4 q[2]; } u;
  u.q[0] = *(const u32x4*)(p);
  u.q[1] = *(const u32x4*)(p + 8);
  return u.v;
}
// B (16-bit, 32x16): lane L holds col N=L&15; K = (L>=16?16:0)+j, j=0..15 -> 32 contiguous bytes.
static __device__ __forceinline__ v16bf ldfragB(const bf16_t* p) {
  union { v16bf v; u32x4 q[2]; } u;
  u.q[0] = *(const u32x4*)(p);
  u.q[1] = *(const u32x4*)(p + 8);
  return u.v;
}
static __device__ __forceinline__ v8f wmma_bf16(v16bf a, v16bf b, v8f c) {
  return __builtin_amdgcn_wmma_f32_16x16x32_bf16(false, a, false, b, (short)0, c, false, false);
}

// ---- prep kernels: f32 -> bf16 weight transpose, bias table expansion ----
__global__ void prep_wqkv(const float* __restrict__ w, bf16_t* __restrict__ wt) {
  int i = blockIdx.x * 256 + threadIdx.x;
  if (i < DIM * DIM3) {
    int k = i / DIM3, n = i - k * DIM3;
    wt[(size_t)n * DIM + k] = (bf16_t)w[i];        // [K][N] -> [N][K]
  }
}
__global__ void prep_wproj(const float* __restrict__ w, bf16_t* __restrict__ wt) {
  int i = blockIdx.x * 256 + threadIdx.x;
  if (i < DIM * DIM) {
    int k = i / DIM, n = i - k * DIM;
    wt[(size_t)n * DIM + k] = (bf16_t)w[i];
  }
}
__global__ void prep_bias(const float* __restrict__ tab, const int* __restrict__ idx,
                          float* __restrict__ o) {
  int i = blockIdx.x * 256 + threadIdx.x;
  if (i < HEADS * TOK * TOK) {
    int h = i / (TOK * TOK);
    int qn = i - h * TOK * TOK;
    o[i] = tab[idx[qn] * HEADS + h];               // [H][49][49]
  }
}

// ---- fully fused window attention: 1 block = 1 window (49 tokens), 8 waves ----
__global__ __launch_bounds__(256)
void swin_attn_fused(const float* __restrict__ x,
                     const bf16_t* __restrict__ wqkv_t,   // [1152][384] bf16
                     const float* __restrict__ qkv_b,     // [1152]
                     const bf16_t* __restrict__ wproj_t,  // [384][384] bf16 (transposed)
                     const float* __restrict__ proj_b,    // [384]
                     const float* __restrict__ bias_full, // [12][49][49]
                     float* __restrict__ out)
{
  __shared__ __attribute__((aligned(16))) bf16_t x_s[MP * SX];   // x window, bf16
  __shared__ __attribute__((aligned(16))) bf16_t q_s[MP * SX];   // q; later reused for O
  __shared__ __attribute__((aligned(16))) bf16_t k_s[MP * SX];   // k
  __shared__ __attribute__((aligned(16))) bf16_t vt_s[DIM * SV]; // v^T: [channel][token]
  __shared__ __attribute__((aligned(16))) bf16_t p_s[HEADS * MP * SP]; // probs; also f32 TDM stage

  const int tid     = threadIdx.x;
  const int wave    = tid >> 5;
  const int lane    = tid & 31;
  const int colb    = lane & 15;          // N (or M for A-frags) within 16-tile
  const int rowhalf = (lane >> 4) & 1;    // lane-half selects K/M sub-block
  const int w       = blockIdx.x;         // window id

  // ---------- Phase 0: TDM DMA of x window (49x384 f32, contiguous) into LDS staging ----------
  float* xf = (float*)p_s;                // 75,264 B staging inside p_s (dead until phase 3)
  if (wave == 0) {
    unsigned long long ga = (unsigned long long)(x + (size_t)w * TOK * DIM);
    unsigned lds_off = (unsigned)(size_t)(void*)xf;
    // D# group0: count=1 | lds_addr | global_addr[56:0] | type=2
    u32x4 g0 = { 1u, lds_off, (u32)ga, (u32)((ga >> 32) & 0x01FFFFFFu) | 0x80000000u };
    // D# group1: data_size=4B; tensor 384x49; tile 384x49; dim0_stride=384 (data_size units)
    i32x8 g1 = { (int)0x20000,                 // workgroup_mask=0, data_size=2 (4B)
                 (int)((DIM & 0xFFFF) << 16),  // tensor_dim0 lo16 in [63:48]
                 (int)(((u32)TOK & 0xFFFFu) << 16) | (DIM >> 16), // dim0 hi16 | tensor_dim1 lo16
                 (int)((DIM << 16) | (TOK >> 16)),                // tensor_dim1 hi16 | tile_dim0
                 (int)TOK,                     // tile_dim1 (tile_dim2=0)
                 (int)DIM,                     // tensor_dim0_stride lo32
                 0, 0 };                       // stride hi bits / dim1_stride unused (2D)
    i32x4 gz = { 0, 0, 0, 0 };
    i32x8 gz8 = { 0, 0, 0, 0, 0, 0, 0, 0 };
    __builtin_amdgcn_tensor_load_to_lds(g0, g1, gz, gz, gz8, 0);
  }
  // prefetch the weight rows this wave touches first (L2 warm-up)
  __builtin_prefetch(wqkv_t + (size_t)(wave * 16 + colb) * DIM, 0, 3);
  __builtin_prefetch(wproj_t + (size_t)(wave * 16 + colb) * DIM, 0, 3);
  if (wave == 0) __builtin_amdgcn_s_wait_tensorcnt(0);
  __syncthreads();

  // ---------- Phase 1: convert staged f32 -> bf16, zero-pad rows 49..63 ----------
  for (int idx = tid; idx < MP * DIM; idx += 256) {
    int t = idx / DIM, c = idx - t * DIM;
    float v = (t < TOK) ? xf[t * DIM + c] : 0.0f;
    x_s[t * SX + c] = (bf16_t)v;
  }
  __syncthreads();

  // ---------- Phase 2: QKV GEMM [64x384]x[384x1152] + bias; q scaled; v stored transposed ----------
  for (int nt = wave; nt < DIM3 / 16; nt += 8) {
    const bf16_t* wrow = wqkv_t + (size_t)(nt * 16 + colb) * DIM + rowhalf * 16;
    v16bf bcur = ldfragB(wrow);
    v8f z = {0.f,0.f,0.f,0.f,0.f,0.f,0.f,0.f};
    v8f acc[4] = {z, z, z, z};
#pragma unroll
    for (int kk = 0; kk < DIM / 32; ++kk) {
      int kn = (kk + 1 < DIM / 32) ? (kk + 1) : 0;
      v16bf bnxt = ldfragB(wrow + kn * 32);          // double-buffer: issue before compute
#pragma unroll
      for (int mt = 0; mt < 4; ++mt) {
        const bf16_t* ap = x_s + (mt * 16 + colb) * SX + kk * 32 + rowhalf * 16;
        acc[mt] = wmma_bf16(ldfragA(ap), bcur, acc[mt]);
      }
      bcur = bnxt;
    }
    int ch  = nt * 16 + colb;            // 0..1151 (uniform section per tile)
    int sec = ch / DIM;                  // 0=q 1=k 2=v
    int c   = ch - sec * DIM;
    float bb = qkv_b[ch];
#pragma unroll
    for (int mt = 0; mt < 4; ++mt) {
      if (sec == 2) {                    // packed 16B store into v^T (tokens contiguous)
        union { u32x4 q; __bf16 b[8]; } pk;
#pragma unroll
        for (int i = 0; i < 8; ++i) pk.b[i] = (bf16_t)(acc[mt][i] + bb);
        *(u32x4*)(&vt_s[c * SV + mt * 16 + rowhalf * 8]) = pk.q;
      } else {
#pragma unroll
        for (int i = 0; i < 8; ++i) {
          int r = mt * 16 + rowhalf * 8 + i;
          float val = acc[mt][i] + bb;
          if (sec == 0) q_s[r * SX + c] = (bf16_t)(val * QSCALE);
          else          k_s[r * SX + c] = (bf16_t)val;
        }
      }
    }
  }
  __syncthreads();

  // ---------- Phase 3: S = q k^T (+bias), masked softmax, P -> LDS (bf16) ----------
  for (int j = 0; j < 6; ++j) {
    int it = wave + 8 * j;               // 48 items = 12 heads x 4 row-blocks
    int h = it >> 2, mt = it & 3;
    const bf16_t* ap = q_s + (mt * 16 + colb) * SX + h * HD + rowhalf * 16;
    v16bf a = ldfragA(ap);
    v8f s[4];
#pragma unroll
    for (int nt = 0; nt < 4; ++nt) {
      const bf16_t* bp = k_s + (nt * 16 + colb) * SX + h * HD + rowhalf * 16;
      v8f z = {0.f,0.f,0.f,0.f,0.f,0.f,0.f,0.f};
      s[nt] = wmma_bf16(a, ldfragB(bp), z);
    }
#pragma unroll
    for (int i = 0; i < 8; ++i) {
      int r = mt * 16 + rowhalf * 8 + i;
      float vals[4];
#pragma unroll
      for (int nt = 0; nt < 4; ++nt) {
        int n = nt * 16 + colb;
        float v = s[nt][i];
        if (n < TOK && r < TOK) v += bias_full[(h * TOK + r) * TOK + n];
        if (n >= TOK) v = -1e30f;        // mask padded keys
        vals[nt] = v;
      }
      float m = fmaxf(fmaxf(vals[0], vals[1]), fmaxf(vals[2], vals[3]));
#pragma unroll
      for (int off = 8; off >= 1; off >>= 1) m = fmaxf(m, __shfl_xor(m, off, 32));
      float sum = 0.f;
#pragma unroll
      for (int nt = 0; nt < 4; ++nt) { vals[nt] = __expf(vals[nt] - m); sum += vals[nt]; }
#pragma unroll
      for (int off = 8; off >= 1; off >>= 1) sum += __shfl_xor(sum, off, 32);
      float inv = 1.0f / sum;
#pragma unroll
      for (int nt = 0; nt < 4; ++nt)
        p_s[(h * MP + r) * SP + nt * 16 + colb] = (bf16_t)(vals[nt] * inv);
    }
  }
  __syncthreads();

  // ---------- Phase 4: O = P V  (per head: M=64, N=32, K=64); O overwrites q_s ----------
  for (int j = 0; j < 12; ++j) {
    int tt = wave + 8 * j;               // 96 tiles
    int h = tt >> 3, sub = tt & 7, mt = sub >> 1, nd = sub & 1;
    v8f acc = {0.f,0.f,0.f,0.f,0.f,0.f,0.f,0.f};
#pragma unroll
    for (int kk = 0; kk < 2; ++kk) {
      const bf16_t* ap = p_s + (h * MP + mt * 16 + colb) * SP + kk * 32 + rowhalf * 16;
      const bf16_t* bp = vt_s + (h * HD + nd * 16 + colb) * SV + kk * 32 + rowhalf * 16;
      acc = wmma_bf16(ldfragA(ap), ldfragB(bp), acc);
    }
    int c = h * HD + nd * 16 + colb;
#pragma unroll
    for (int i = 0; i < 8; ++i)
      q_s[(mt * 16 + rowhalf * 8 + i) * SX + c] = (bf16_t)acc[i];
  }
  __syncthreads();

  // ---------- Phase 5: out = O proj_w + proj_b ----------
  for (int nt = wave; nt < DIM / 16; nt += 8) {
    const bf16_t* wrow = wproj_t + (size_t)(nt * 16 + colb) * DIM + rowhalf * 16;
    v16bf bcur = ldfragB(wrow);
    v8f z = {0.f,0.f,0.f,0.f,0.f,0.f,0.f,0.f};
    v8f acc[4] = {z, z, z, z};
#pragma unroll
    for (int kk = 0; kk < DIM / 32; ++kk) {
      int kn = (kk + 1 < DIM / 32) ? (kk + 1) : 0;
      v16bf bnxt = ldfragB(wrow + kn * 32);
#pragma unroll
      for (int mt = 0; mt < 4; ++mt) {
        const bf16_t* ap = q_s + (mt * 16 + colb) * SX + kk * 32 + rowhalf * 16;
        acc[mt] = wmma_bf16(ldfragA(ap), bcur, acc[mt]);
      }
      bcur = bnxt;
    }
    int ch = nt * 16 + colb;
    float pb = proj_b[ch];
#pragma unroll
    for (int mt = 0; mt < 4; ++mt) {
#pragma unroll
      for (int i = 0; i < 8; ++i) {
        int r = mt * 16 + rowhalf * 8 + i;
        if (r < TOK)
          out[((size_t)w * TOK + r) * DIM + ch] = acc[mt][i] + pb;
      }
    }
  }
}

extern "C" void kernel_launch(void* const* d_in, const int* in_sizes, int n_in,
                              void* d_out, int out_size, void* d_ws, size_t ws_size,
                              hipStream_t stream) {
  const float* x       = (const float*)d_in[0];
  const float* qkv_w   = (const float*)d_in[1];
  const float* qkv_b   = (const float*)d_in[2];
  const float* proj_w  = (const float*)d_in[3];
  const float* proj_b  = (const float*)d_in[4];
  const float* rel_tab = (const float*)d_in[5];
  const int*   rel_idx = (const int*)d_in[6];
  float* out = (float*)d_out;

  // workspace layout: bf16 qkv_w^T | bf16 proj_w^T | f32 expanded bias
  bf16_t* wqkv_t  = (bf16_t*)d_ws;
  bf16_t* wproj_t = wqkv_t + (size_t)DIM * DIM3;
  float*  bias_f  = (float*)(wproj_t + (size_t)DIM * DIM);

  prep_wqkv <<<(DIM * DIM3 + 255) / 256, 256, 0, stream>>>(qkv_w, wqkv_t);
  prep_wproj<<<(DIM * DIM  + 255) / 256, 256, 0, stream>>>(proj_w, wproj_t);
  prep_bias <<<(HEADS * TOK * TOK + 255) / 256, 256, 0, stream>>>(rel_tab, rel_idx, bias_f);

  int nwin = in_sizes[0] / (TOK * DIM);   // 2048
  swin_attn_fused<<<nwin, 256, 0, stream>>>(x, wqkv_t, qkv_b, wproj_t, proj_b, bias_f, out);
}
